// att_classifier_49563922596466
// MI455X (gfx1250) — compile-verified, matrix operand order checked by reading, not run on hardware
//
#include <hip/hip_runtime.h>
#include <hip/hip_bf16.h>
#include <math.h>

#define D   768
#define NH  12
#define HD  64
#define DFF 3072
#define NL  4
#define Bsz 4
#define S   512
#define Hh  32
#define Qq  64
#define Vv  16
#define BS  (Bsz * S)

typedef __attribute__((ext_vector_type(16))) __bf16 v16bf;
typedef __attribute__((ext_vector_type(8)))  float  v8f;
typedef unsigned short u16;
typedef unsigned int   u32;

union AFrag { u32 u[8]; v16bf v; };

__device__ __forceinline__ u32 pk2bf(float a, float b) {
    u32 ua = __float_as_uint(a), ub = __float_as_uint(b);
    ua += 0x7FFFu + ((ua >> 16) & 1u);           // RNE to bf16
    ub += 0x7FFFu + ((ub >> 16) & 1u);
    return (ua >> 16) | (ub & 0xFFFF0000u);
}
__device__ __forceinline__ u16 f2bf(float a) {
    u32 u = __float_as_uint(a);
    u += 0x7FFFu + ((u >> 16) & 1u);
    return (u16)(u >> 16);
}

// ---------------------------------------------------------------------------
// Weight repack: f32 [K,N] row-major -> fragment-major bf16.
// Layout: out[((tn*(K/32)+kc)*32 + lane)*8 + i], so each lane's B fragment for
// a (tn, kc) tile is 32 contiguous bytes (-> global_load_b128 in the GEMM).
// ---------------------------------------------------------------------------
__global__ void __launch_bounds__(256) pack_w(
    const float* __restrict__ W, u32* __restrict__ out, int N, int K, int total)
{
    int kcN = K >> 5;
    for (int e = blockIdx.x * 256 + threadIdx.x; e < total; e += gridDim.x * 256) {
        int i = e & 7, lane = (e >> 3) & 31, chunk = e >> 8;
        int kc = chunk % kcN, tn = chunk / kcN;
        int kbase = (lane >> 4) << 3;
        int kk = (i < 4) ? (kbase + 2 * i) : (16 + kbase + 2 * (i - 4));
        int k = kc * 32 + kk;
        int n = tn * 16 + (lane & 15);
        out[e] = pk2bf(W[(size_t)k * N + n], W[(size_t)(k + 1) * N + n]);
    }
}

// flat f32 -> nothing needed: activations are converted at producer kernels.

// ---------------------------------------------------------------------------
// WMMA GEMM: C[M,N] = act(A[M,K] @ W[K,N] + bias).
// A: bf16 row-major.  W: fragment-major packed bf16.  Output f32 and/or bf16.
// One 64(M) x 16(N) block per wave (4 M-tiles share one weight fragment).
// Requires M % 64 == 0, N % 16 == 0, K % 32 == 0.
// ---------------------------------------------------------------------------
__global__ void __launch_bounds__(256) gemm_bf16_wmma(
    const u16* __restrict__ A, const u32* __restrict__ Wpk,
    const float* __restrict__ bias, float* __restrict__ Cf,
    u16* __restrict__ Cbf, int M, int N, int K, int act)
{
    const int lane  = threadIdx.x & 31;
    const int wid   = blockIdx.x * (blockDim.x >> 5) + (threadIdx.x >> 5);
    const int tilesN = N >> 4;
    const int tiles  = (M >> 6) * tilesN;
    if (wid >= tiles) return;                    // wave-uniform exit
    const int tm = wid / tilesN, tn = wid % tilesN;
    const int mrow = lane & 15;
    const int kb   = (lane >> 4) << 3;
    const int kchunks = K >> 5;

    v8f acc[4];
#pragma unroll
    for (int t = 0; t < 4; ++t)
#pragma unroll
        for (int r = 0; r < 8; ++r) acc[t][r] = 0.f;

    const u16* a0    = A + (size_t)(tm * 64 + mrow) * K;
    const u32* bbase = Wpk + (size_t)tn * kchunks * 256 + lane * 8;

    for (int kc = 0; kc < kchunks; ++kc) {
        const u32* bp = bbase + (size_t)kc * 256;
        if (kc + 1 < kchunks) {                  // CDNA5 global_prefetch_b8
            __builtin_prefetch(bp + 256, 0, 1);
            __builtin_prefetch(a0 + kc * 32 + 32, 0, 1);
        }
        AFrag b;
#pragma unroll
        for (int i = 0; i < 8; ++i) b.u[i] = bp[i];      // 2x b128, coalesced
#pragma unroll
        for (int t = 0; t < 4; ++t) {
            AFrag a;
            const u16* ap = a0 + (size_t)(t * 16) * K + kc * 32;
#pragma unroll
            for (int i = 0; i < 4; ++i) {
                a.u[i]     = *(const u32*)(ap + kb + 2 * i);        // 16B contig
                a.u[4 + i] = *(const u32*)(ap + 16 + kb + 2 * i);   // 16B contig
            }
            acc[t] = __builtin_amdgcn_wmma_f32_16x16x32_bf16(
                         false, a.v, false, b.v, (short)0, acc[t], false, false);
        }
    }

    const int nq = tn * 16 + mrow;
    const float bv = bias ? bias[nq] : 0.f;
#pragma unroll
    for (int t = 0; t < 4; ++t)
#pragma unroll
        for (int r = 0; r < 8; ++r) {
            size_t idx = (size_t)(tm * 64 + t * 16 + r + ((lane >> 4) << 3)) * N + nq;
            float c = acc[t][r] + bv;
            if (act == 1) c = 0.5f * c * (1.f + erff(c * 0.70710678118654752f));
            if (Cf)  Cf[idx]  = c;
            if (Cbf) Cbf[idx] = f2bf(c);
        }
}

// ---------------------------------------------------------------------------
// Flash attention, bf16 in / bf16 out.  One wave per (b, h, 16-q tile).
// ---------------------------------------------------------------------------
__global__ void __launch_bounds__(256) flash_attn(
    const u16* __restrict__ Qb, const u16* __restrict__ Kb,
    const u16* __restrict__ Vb, const int* __restrict__ amask,
    u16* __restrict__ Ob)
{
    __shared__ u16 pb[8][16 * 32];               // per-wave P staging (bf16)
    const int lane  = threadIdx.x & 31;
    const int wslot = threadIdx.x >> 5;
    const int wid   = blockIdx.x * 8 + wslot;
    const int qt = wid & 31;
    const int h  = (wid >> 5) % NH;
    const int b  = wid / (32 * NH);

    const int mrow = lane & 15;
    const int kb   = (lane >> 4) << 3;
    const int hi   = lane >> 4;

    AFrag aq[2];
    {
        const u16* qp = Qb + (size_t)(b * S + qt * 16 + mrow) * D + h * HD;
#pragma unroll
        for (int c = 0; c < 2; ++c) {
            const u16* qc = qp + c * 32;
#pragma unroll
            for (int i = 0; i < 4; ++i) {
                aq[c].u[i]     = *(const u32*)(qc + kb + 2 * i);
                aq[c].u[4 + i] = *(const u32*)(qc + 16 + kb + 2 * i);
            }
        }
    }

    v8f o[4];
    float rowm[8], rowl[8];
#pragma unroll
    for (int t = 0; t < 4; ++t)
#pragma unroll
        for (int r = 0; r < 8; ++r) o[t][r] = 0.f;
#pragma unroll
    for (int r = 0; r < 8; ++r) { rowm[r] = -3.0e38f; rowl[r] = 0.f; }

    u16* pbw = pb[wslot];

    for (int kc = 0; kc < S / 32; ++kc) {
        v8f s0 = {0.f,0.f,0.f,0.f,0.f,0.f,0.f,0.f};
        v8f s1 = {0.f,0.f,0.f,0.f,0.f,0.f,0.f,0.f};
#pragma unroll
        for (int c = 0; c < 2; ++c) {
            AFrag bk0, bk1;
            const u16* kp0 = Kb + (size_t)(b * S + kc * 32 + mrow) * D + h * HD + c * 32;
            const u16* kp1 = kp0 + (size_t)16 * D;
#pragma unroll
            for (int i = 0; i < 4; ++i) {
                bk0.u[i]     = *(const u32*)(kp0 + kb + 2 * i);
                bk0.u[4 + i] = *(const u32*)(kp0 + 16 + kb + 2 * i);
                bk1.u[i]     = *(const u32*)(kp1 + kb + 2 * i);
                bk1.u[4 + i] = *(const u32*)(kp1 + 16 + kb + 2 * i);
            }
            s0 = __builtin_amdgcn_wmma_f32_16x16x32_bf16(false, aq[c].v, false, bk0.v, (short)0, s0, false, false);
            s1 = __builtin_amdgcn_wmma_f32_16x16x32_bf16(false, aq[c].v, false, bk1.v, (short)0, s1, false, false);
        }
        const int key0 = kc * 32 + mrow;
        const float am0 = amask[b * S + key0]      ? 0.f : -1.0e9f;
        const float am1 = amask[b * S + key0 + 16] ? 0.f : -1.0e9f;

#pragma unroll
        for (int r = 0; r < 8; ++r) {
            float v0 = s0[r] * 0.125f + am0;
            float v1 = s1[r] * 0.125f + am1;
            float mx = fmaxf(v0, v1);
#pragma unroll
            for (int off = 8; off; off >>= 1) mx = fmaxf(mx, __shfl_xor(mx, off, 16));
            float nm = fmaxf(rowm[r], mx);
            float sc = expf(rowm[r] - nm);
            float p0 = expf(v0 - nm);
            float p1 = expf(v1 - nm);
            float rs = p0 + p1;
#pragma unroll
            for (int off = 8; off; off >>= 1) rs += __shfl_xor(rs, off, 16);
            rowm[r] = nm;
            rowl[r] = rowl[r] * sc + rs;
#pragma unroll
            for (int t = 0; t < 4; ++t) o[t][r] *= sc;
            int mr = r + (hi << 3);
            pbw[mr * 32 + mrow]      = f2bf(p0);
            pbw[mr * 32 + 16 + mrow] = f2bf(p1);
        }
        asm volatile("s_wait_dscnt 0x0" ::: "memory");

        AFrag apf;
#pragma unroll
        for (int i = 0; i < 4; ++i) {
            apf.u[i]     = *(const u32*)&pbw[mrow * 32 + kb + 2 * i];
            apf.u[4 + i] = *(const u32*)&pbw[mrow * 32 + 16 + kb + 2 * i];
        }
#pragma unroll
        for (int t = 0; t < 4; ++t) {
            AFrag bvv;
            const u16* vp = Vb + (size_t)(b * S + kc * 32) * D + h * HD + t * 16 + mrow;
#pragma unroll
            for (int i = 0; i < 4; ++i) {
                bvv.u[i]     = (u32)vp[(size_t)(kb + 2*i) * D]      | ((u32)vp[(size_t)(kb + 2*i + 1) * D] << 16);
                bvv.u[4 + i] = (u32)vp[(size_t)(16 + kb + 2*i) * D] | ((u32)vp[(size_t)(16 + kb + 2*i + 1) * D] << 16);
            }
            o[t] = __builtin_amdgcn_wmma_f32_16x16x32_bf16(false, apf.v, false, bvv.v, (short)0, o[t], false, false);
        }
    }

#pragma unroll
    for (int t = 0; t < 4; ++t)
#pragma unroll
        for (int r = 0; r < 8; ++r) {
            int mq = qt * 16 + r + (hi << 3);
            Ob[(size_t)(b * S + mq) * D + h * HD + t * 16 + mrow] = f2bf(o[t][r] / rowl[r]);
        }
}

// ---------------------------------------------------------------------------
// Embedding gather + LayerNorm (eps 1e-12).  f32 out + bf16 mirror.
// ---------------------------------------------------------------------------
__global__ void __launch_bounds__(256) embed_ln(
    const int* __restrict__ x, const float* __restrict__ ew,
    const float* __restrict__ ep, const float* __restrict__ et,
    const float* __restrict__ g, const float* __restrict__ bb,
    float* __restrict__ out, u16* __restrict__ obf)
{
    __shared__ float red[256];
    const int row = blockIdx.x;
    const int s   = row % S;
    const float* w = ew + (size_t)x[row] * D;
    const float* p = ep + (size_t)s * D;
    float v[3]; float sum = 0.f;
    for (int i = threadIdx.x, j = 0; i < D; i += 256, ++j) { v[j] = w[i] + p[i] + et[i]; sum += v[j]; }
    red[threadIdx.x] = sum; __syncthreads();
    for (int st = 128; st; st >>= 1) { if (threadIdx.x < st) red[threadIdx.x] += red[threadIdx.x + st]; __syncthreads(); }
    float mean = red[0] / (float)D; __syncthreads();
    float vs = 0.f;
    for (int i = threadIdx.x, j = 0; i < D; i += 256, ++j) { float d_ = v[j] - mean; vs += d_ * d_; }
    red[threadIdx.x] = vs; __syncthreads();
    for (int st = 128; st; st >>= 1) { if (threadIdx.x < st) red[threadIdx.x] += red[threadIdx.x + st]; __syncthreads(); }
    float rstd = rsqrtf(red[0] / (float)D + 1e-12f);
    for (int i = threadIdx.x, j = 0; i < D; i += 256, ++j) {
        float y = (v[j] - mean) * rstd * g[i] + bb[i];
        out[(size_t)row * D + i] = y;
        obf[(size_t)row * D + i] = f2bf(y);
    }
}

// ---------------------------------------------------------------------------
// out = LayerNorm(X + R); f32 out (+ optional bf16 mirror).  Safe in-place.
// ---------------------------------------------------------------------------
__global__ void __launch_bounds__(256) add_ln(
    const float* __restrict__ X, const float* __restrict__ R,
    const float* __restrict__ g, const float* __restrict__ bb,
    float* __restrict__ out, u16* __restrict__ obf, float eps)
{
    __shared__ float red[256];
    const int row = blockIdx.x;
    float v[3]; float sum = 0.f;
    for (int i = threadIdx.x, j = 0; i < D; i += 256, ++j) {
        v[j] = X[(size_t)row * D + i] + R[(size_t)row * D + i]; sum += v[j];
    }
    red[threadIdx.x] = sum; __syncthreads();
    for (int st = 128; st; st >>= 1) { if (threadIdx.x < st) red[threadIdx.x] += red[threadIdx.x + st]; __syncthreads(); }
    float mean = red[0] / (float)D; __syncthreads();
    float vs = 0.f;
    for (int i = threadIdx.x, j = 0; i < D; i += 256, ++j) { float d_ = v[j] - mean; vs += d_ * d_; }
    red[threadIdx.x] = vs; __syncthreads();
    for (int st = 128; st; st >>= 1) { if (threadIdx.x < st) red[threadIdx.x] += red[threadIdx.x + st]; __syncthreads(); }
    float rstd = rsqrtf(red[0] / (float)D + eps);
    for (int i = threadIdx.x, j = 0; i < D; i += 256, ++j) {
        float y = (v[j] - mean) * rstd * g[i] + bb[i];
        out[(size_t)row * D + i] = y;
        if (obf) obf[(size_t)row * D + i] = f2bf(y);
    }
}

// hs[b, sel[b,i], :] -> out[b*per_b + i, :]  (f32)
__global__ void __launch_bounds__(256) gather_rows(
    const float* __restrict__ hs, const int* __restrict__ sel,
    float* __restrict__ out, int per_b)
{
    int row = blockIdx.x; int b = row / per_b; int i = row % per_b;
    const float* sp = hs + (size_t)(b * S + sel[b * per_b + i]) * D;
    float* op = out + (size_t)row * D;
    for (int t = threadIdx.x; t < D; t += blockDim.x) op[t] = sp[t];
}

// head_part[b, i, :] -> hq[(b*4+j)*32 + i, :]  f32 + bf16 mirror
__global__ void __launch_bounds__(256) repeat_h(
    const float* __restrict__ hp, float* __restrict__ hq, u16* __restrict__ hqbf)
{
    int row = blockIdx.x;            // 0..511
    int b = row >> 7;
    int i = row & 31;
    const float* sp = hp + (size_t)(b * Hh + i) * D;
    for (int t = threadIdx.x; t < D; t += blockDim.x) {
        float y = sp[t];
        hq[(size_t)row * D + t]   = y;
        hqbf[(size_t)row * D + t] = f2bf(y);
    }
}

// key[(b*4+j)*16 + vk, :] = valid ? qu_part[b, span_idx, :] : 0   (bf16)
__global__ void __launch_bounds__(256) build_key(
    const float* __restrict__ qp, const int* __restrict__ spi,
    const int* __restrict__ spl, u16* __restrict__ key)
{
    int row = blockIdx.x;            // 0..255
    int vk = row & 15; int bb = row >> 4; int b = bb >> 2; int j = bb & 3;
    int idx = spi[(b * 4 + j) * Vv + vk];
    bool valid = vk < spl[b * 4 + j];
    const float* sp = qp + (size_t)(b * Qq + idx) * D;
    u16* op = key + (size_t)row * D;
    for (int t = threadIdx.x; t < D; t += blockDim.x) op[t] = f2bf(valid ? sp[t] : 0.f);
}

// cross attention (scalar, tiny): q [16,32,D] f32, k=v [16,16,D] f32 -> bf16 out
__global__ void __launch_bounds__(32) cross_attn(
    const float* __restrict__ cq, const float* __restrict__ ck,
    const float* __restrict__ cv, const int* __restrict__ spl,
    u16* __restrict__ catt)
{
    int bh = blockIdx.x;             // 0..191
    int h = bh % NH; int bb = bh / NH;
    int b = bb >> 2; int j = bb & 3;
    int len = spl[b * 4 + j];
    int qrow = threadIdx.x;
    const float* qv = cq + (size_t)(bb * Hh + qrow) * D + h * HD;
    float sc[16]; float mx = -3.0e38f;
    for (int k = 0; k < 16; ++k) {
        float s = 0.f;
        const float* kv = ck + (size_t)(bb * 16 + k) * D + h * HD;
        for (int d0 = 0; d0 < HD; ++d0) s += qv[d0] * kv[d0];
        s *= 0.125f;
        if (!((k < len) || (k == 0))) s = -1.0e9f;
        sc[k] = s; mx = fmaxf(mx, s);
    }
    float den = 0.f;
    for (int k = 0; k < 16; ++k) { sc[k] = expf(sc[k] - mx); den += sc[k]; }
    float inv = 1.f / den;
    float o[HD];
    for (int d0 = 0; d0 < HD; ++d0) o[d0] = 0.f;
    for (int k = 0; k < 16; ++k) {
        const float* vv = cv + (size_t)(bb * 16 + k) * D + h * HD;
        float a = sc[k] * inv;
        for (int d0 = 0; d0 < HD; ++d0) o[d0] += a * vv[d0];
    }
    u16* op = catt + (size_t)(bb * Hh + qrow) * D + h * HD;
    for (int d0 = 0; d0 < HD; ++d0) op[d0] = f2bf(o[d0]);
}

// out[row, :] = log_softmax(X[row, :] @ W[D, C] + b).  One wave per row.
__global__ void __launch_bounds__(32) logsm(
    const float* __restrict__ X, const float* __restrict__ W,
    const float* __restrict__ bb, float* __restrict__ out, int C)
{
    int row = blockIdx.x;
    int c = threadIdx.x;
    float lg = 0.f;
    if (c < C) {
        float s = bb[c];
        const float* xp = X + (size_t)row * D;
        for (int d0 = 0; d0 < D; ++d0) s += xp[d0] * W[d0 * C + c];
        lg = s;
    }
    float m = (c < C) ? lg : -3.0e38f;
    for (int off = 16; off; off >>= 1) m = fmaxf(m, __shfl_xor(m, off, 32));
    float e = (c < C) ? expf(lg - m) : 0.f;
    for (int off = 16; off; off >>= 1) e += __shfl_xor(e, off, 32);
    if (c < C) out[(size_t)row * C + c] = lg - m - logf(e);
}

// ---------------------------------------------------------------------------
extern "C" void kernel_launch(void* const* d_in, const int* in_sizes, int n_in,
                              void* d_out, int out_size, void* d_ws, size_t ws_size,
                              hipStream_t stream) {
    (void)in_sizes; (void)n_in; (void)out_size; (void)ws_size;
    auto F = [&](int i) { return (const float*)d_in[i]; };
    auto I = [&](int i) { return (const int*)d_in[i]; };

    // depth-first insertion-order flattening of setup_inputs():
    // 0..4 emb_word, emb_pos, emb_type, emb_ln_g, emb_ln_b
    // 5+16l: Wq,bq,Wk,bk,Wv,bv,Wo,bo,ln1_g,ln1_b,W1,b1,W2,b2,ln2_g,ln2_b (l=0..3)
    // 69 Wsel 70 bsel 71 Wcon 72 bcon
    // 73..80 cross Wq,bq,Wk,bk,Wv,bv,Wo,bo
    // 81 ln_g 82 ln_b 83 f1W 84 f1b 85 f2W 86 f2b 87 Wtwo 88 btwo
    // 89 x 90 attention_mask 91 qu_sel 92 head_sel 93 span_indices 94 span_len 95 head_len

    const size_t R = (size_t)BS * D;             // 1,572,864

    // ---- f32 region
    float* ws   = (float*)d_ws;
    float* hs   = ws;                            // [2048,768]
    float* bufP = ws + R;                        // [2048,768]
    float* hq   = ws + 2 * R;                    // [512,768]
    float* cqb  = hq   + (size_t)512 * D;
    float* ckb  = cqb  + (size_t)512 * D;        // [256,768]
    float* cvb  = ckb  + (size_t)256 * D;        // [256,768]
    float* cprj = cvb  + (size_t)256 * D;        // [512,768]
    float* zbf  = cprj + (size_t)512 * D;        // [512,768]
    float* cf2  = zbf  + (size_t)512 * D;        // [512,768]
    float* pch  = cf2  + (size_t)512 * D;        // [512,768]
    float* f32end = pch + (size_t)512 * D;

    // ---- bf16 region
    u16* hsb  = (u16*)f32end;                    // [2048,768]
    u16* Qbf  = hsb  + R;
    u16* Kbf  = Qbf  + R;
    u16* Vbf  = Kbf  + R;
    u16* Obf  = Vbf  + R;
    u16* Fbf  = Obf  + R;                        // [2048,3072]
    u16* hqb  = Fbf  + (size_t)BS * DFF;         // [512,768]
    u16* keyb = hqb  + (size_t)512 * D;          // [256,768]
    u16* cab  = keyb + (size_t)256 * D;          // [512,768]  (cross attn out)
    u16* zbb  = cab  + (size_t)512 * D;          // [512,768]
    u16* cf1b = zbb  + (size_t)512 * D;          // [512,2048]
    u16* bf16end = cf1b + (size_t)512 * 2048;

    // ---- packed-weight region (u32)
    u32* wp = (u32*)bf16end;
    const size_t sDD = (size_t)D * D / 2;        // 294,912 u32
    const size_t sDF = (size_t)D * DFF / 2;      // 1,179,648 u32
    const size_t sD2k = (size_t)D * 2048 / 2;    // 786,432 u32
    u32* pW[NL][6];
    for (int l = 0; l < NL; ++l) {
        for (int t = 0; t < 4; ++t) { pW[l][t] = wp; wp += sDD; }   // Wq,Wk,Wv,Wo
        pW[l][4] = wp; wp += sDF;                                    // W1
        pW[l][5] = wp; wp += sDF;                                    // W2
    }
    u32* pC[4];
    for (int t = 0; t < 4; ++t) { pC[t] = wp; wp += sDD; }           // cross Wq..Wo
    u32* pF1 = wp; wp += sD2k;
    u32* pF2 = wp; wp += sD2k;

    float* out       = (float*)d_out;
    float* p_head    = out;                         // [4,32,7]   = 896
    float* p_qu      = out + 896;                   // [4,64,4]   = 1024
    float* head_part = out + 1920;                  // [4,32,768] = 98304
    float* qu_part   = out + 1920 + 98304;          // [4,64,768] = 196608
    float* ans       = out + 296832;                // [4,128,2]  = 1024

    auto packw = [&](const float* W, u32* dst, int N_, int K_) {
        int total = (K_ * N_) >> 1;
        pack_w<<<(total + 255) / 256, 256, 0, stream>>>(W, dst, N_, K_, total);
    };
    auto gemm = [&](const u16* A, const u32* Wm, const float* bias,
                    float* Cf, u16* Cbf, int M, int N, int K, int act) {
        int tiles = (M / 64) * (N / 16);
        gemm_bf16_wmma<<<(tiles + 7) / 8, 256, 0, stream>>>(A, Wm, bias, Cf, Cbf, M, N, K, act);
    };

    // ---- pack all weights once (bandwidth-trivial, reused 32x per GEMM)
    for (int l = 0; l < NL; ++l) {
        const int p = 5 + 16 * l;
        packw(F(p + 0),  pW[l][0], D, D);
        packw(F(p + 2),  pW[l][1], D, D);
        packw(F(p + 4),  pW[l][2], D, D);
        packw(F(p + 6),  pW[l][3], D, D);
        packw(F(p + 10), pW[l][4], DFF, D);
        packw(F(p + 12), pW[l][5], D, DFF);
    }
    packw(F(73), pC[0], D, D);  packw(F(75), pC[1], D, D);
    packw(F(77), pC[2], D, D);  packw(F(79), pC[3], D, D);
    packw(F(83), pF1, 2048, D);
    packw(F(85), pF2, D, 2048);

    // ---- embeddings + LN
    embed_ln<<<BS, 256, 0, stream>>>(I(89), F(0), F(1), F(2), F(3), F(4), hs, hsb);

    // ---- 4 BERT layers
    for (int l = 0; l < NL; ++l) {
        const int p = 5 + 16 * l;
        gemm(hsb, pW[l][0], F(p + 1), nullptr, Qbf, BS, D, D, 0);
        gemm(hsb, pW[l][1], F(p + 3), nullptr, Kbf, BS, D, D, 0);
        gemm(hsb, pW[l][2], F(p + 5), nullptr, Vbf, BS, D, D, 0);
        flash_attn<<<(Bsz * NH * (S / 16)) / 8, 256, 0, stream>>>(Qbf, Kbf, Vbf, I(90), Obf);
        gemm(Obf, pW[l][3], F(p + 7), bufP, nullptr, BS, D, D, 0);
        add_ln<<<BS, 256, 0, stream>>>(hs, bufP, F(p + 8), F(p + 9), hs, hsb, 1e-12f);
        gemm(hsb, pW[l][4], F(p + 11), nullptr, Fbf, BS, DFF, D, 1);  // exact GELU
        gemm(Fbf, pW[l][5], F(p + 13), bufP, nullptr, BS, D, DFF, 0);
        add_ln<<<BS, 256, 0, stream>>>(hs, bufP, F(p + 14), F(p + 15), hs, hsb, 1e-12f);
    }

    // ---- gathers + per-token heads
    gather_rows<<<Bsz * Hh, 256, 0, stream>>>(hs, I(92), head_part, Hh);
    gather_rows<<<Bsz * Qq, 256, 0, stream>>>(hs, I(91), qu_part, Qq);
    logsm<<<Bsz * Hh, 32, 0, stream>>>(head_part, F(69), F(70), p_head, 7);
    logsm<<<Bsz * Qq, 32, 0, stream>>>(qu_part,  F(71), F(72), p_qu, 4);

    // ---- cross-attention block
    repeat_h<<<512, 256, 0, stream>>>(head_part, hq, hqb);
    build_key<<<256, 256, 0, stream>>>(qu_part, I(93), I(94), keyb);
    gemm(hqb,  pC[0], F(74), cqb, nullptr, 512, D, D, 0);
    gemm(keyb, pC[1], F(76), ckb, nullptr, 256, D, D, 0);
    gemm(keyb, pC[2], F(78), cvb, nullptr, 256, D, D, 0);
    cross_attn<<<16 * NH, 32, 0, stream>>>(cqb, ckb, cvb, I(94), cab);
    gemm(cab, pC[3], F(80), cprj, nullptr, 512, D, D, 0);
    add_ln<<<512, 256, 0, stream>>>(hq, cprj, F(81), F(82), zbf, zbb, 1e-5f);
    gemm(zbb,  pF1, F(84), nullptr, cf1b, 512, 2048, D, 0);          // no activation
    gemm(cf1b, pF2, F(86), cf2, nullptr, 512, D, 2048, 0);
    add_ln<<<512, 256, 0, stream>>>(hq, cf2, F(81), F(82), pch, nullptr, 1e-5f);
    logsm<<<512, 32, 0, stream>>>(pch, F(87), F(88), ans, 2);        // ans = contiguous lp
}
